// GCN_FA_9560597201075
// MI455X (gfx1250) — matrix operand order repaired and checked by reading, not compiled, not run on hardware
//
#include <hip/hip_runtime.h>
#include <hip/hip_bf16.h>
#include <math.h>

// ---------------------------------------------------------------------------
// GCN forward. The reference broadcasts column sums to every row before
// log_softmax, so only one 40-vector must be computed, then broadcast.
//
//   k_init        : deg=1 (self-loop), agg=0, S=0
//   k_deg         : deg[dst] += ew[e]                  (global f32 atomics)
//   k_dinv        : dinv = deg>0 ? rsqrt(deg) : 0
//   k_gemm_wmma   : h = x @ W1, v_wmma_f32_16x16x32_f16; W1 staged in LDS via
//                   global_load_async_to_lds_b128 (ASYNCcnt) + f16 transpose
//   k_scatter     : agg[dst] += h[src]*norm, wave32 per edge (dominant cost)
//   k_node_reduce : S += relu(agg + h*dinv^2 + b1)     (LDS partial + atomics)
//   k_final       : res = log_softmax(S @ W2 + N*b2)
//   k_bcast       : out[i,:] = res
// ---------------------------------------------------------------------------

typedef __attribute__((ext_vector_type(16))) _Float16 v16h;
typedef __attribute__((ext_vector_type(8)))  float    v8f;

#define Fin 128
#define Hdim 64
#define Cdim 40

__global__ void k_init(float* __restrict__ deg, float* __restrict__ agg,
                       float* __restrict__ S, int N) {
    long long idx = (long long)blockIdx.x * blockDim.x + threadIdx.x;
    if (idx < (long long)N * Hdim) agg[idx] = 0.0f;
    if (idx < N)    deg[idx] = 1.0f;   // self-loop weight
    if (idx < Hdim) S[idx]   = 0.0f;
}

__global__ void k_deg(const long long* __restrict__ dst,
                      const float* __restrict__ ew,
                      float* __restrict__ deg, int E) {
    int e = blockIdx.x * blockDim.x + threadIdx.x;
    if (e >= E) return;
    atomicAdd(&deg[(int)dst[e]], ew[e]);
}

__global__ void k_dinv(const float* __restrict__ deg,
                       float* __restrict__ dinv, int N) {
    int i = blockIdx.x * blockDim.x + threadIdx.x;
    if (i >= N) return;
    float d = deg[i];
    dinv[i] = (d > 0.0f) ? rsqrtf(fmaxf(d, 1e-12f)) : 0.0f;
}

// h[N,64] = x[N,128] @ W1[128,64], one wave32 per 16x16 output tile.
// W1 staged: raw f32 -> LDS via async-to-LDS DMA, then converted to a
// transposed f16 LDS image s_w1h[n][k] so B fragments are contiguous.
// A fragment (16x32 f16): lane L holds row M=L&15; half j -> K=kb+j,
// kb=(L<16)?0:16.                                   (contiguous K order, same as B)
// B fragment (32x16 f16): lane L holds col n=L&15; half j -> K=base+j,
// base=(L<16)?0:16.                                 (contiguous K order, same as A)
// D (8 VGPR f32): element r -> row M=r+((L<16)?0:8), col n=L&15.
__global__ void __launch_bounds__(256)
k_gemm_wmma(const float* __restrict__ x, const float* __restrict__ W1,
            float* __restrict__ h, int N) {
    __shared__ float    s_w1f[Fin * Hdim];      // 32 KB raw f32 staging
    __shared__ _Float16 s_w1h[Hdim][Fin];       // 16 KB transposed f16

    // ---- async DMA: W1 (32 KB) -> LDS, 16 B per lane per issue ----
    {
        unsigned lds_base;
        {
            __attribute__((address_space(3))) float* p3 =
                (__attribute__((address_space(3))) float*)s_w1f;
            lds_base = (unsigned)(unsigned long long)p3;
        }
        unsigned t16  = threadIdx.x * 16u;
        unsigned ldsa = lds_base + t16;
        unsigned goff = t16;
        #pragma unroll
        for (int i = 0; i < (Fin * Hdim * 4) / (256 * 16); ++i) {  // 8 iters
            asm volatile("global_load_async_to_lds_b128 %0, %1, %2 offset:0"
                         :: "v"(ldsa), "v"(goff), "s"(W1) : "memory");
            ldsa += 4096;
            goff += 4096;
        }
        asm volatile("s_wait_asynccnt 0" ::: "memory");
    }
    __syncthreads();

    // ---- convert + transpose: s_w1f[k*64+n] -> s_w1h[n][k] (f16) ----
    for (int t = threadIdx.x; t < Fin * Hdim; t += 256) {
        int k = t >> 6;
        int n = t & 63;
        s_w1h[n][k] = (_Float16)s_w1f[t];
    }
    __syncthreads();

    const int lane = threadIdx.x & 31;
    const int wave = threadIdx.x >> 5;
    const int gw   = blockIdx.x * (blockDim.x >> 5) + wave;  // global wave id
    const int tiles_m = (N + 15) >> 4;
    const int tile_m  = gw >> 2;          // Hdim/16 = 4 tiles along columns
    const int tile_n  = gw & 3;
    if (tile_m >= tiles_m) return;        // whole-wave uniform exit (after syncs)

    const int m      = lane & 15;
    const int hiHalf = lane >> 4;         // 0 or 1
    int row  = tile_m * 16 + m;
    int rowc = row < N ? row : N - 1;     // clamp reads; keep EXEC all-ones
    const float* xr = x + (long long)rowc * Fin;

    const int n   = lane & 15;
    const int kgA = hiHalf * 16;          // A K base
    const int kgB = hiHalf * 16;          // B K base
    const int n2  = tile_n * 16 + n;

    v8f acc = {};
    #pragma unroll
    for (int kk = 0; kk < Fin; kk += 32) {
        v16h a;
        #pragma unroll
        for (int j = 0; j < 8; ++j) {
            a[j]     = (_Float16)xr[kk + kgA + j];
            a[j + 8] = (_Float16)xr[kk + kgA + 8 + j];
        }
        // 16 contiguous halves, 32-B aligned -> 2x ds_load_b128
        v16h b = *(const v16h*)&s_w1h[n2][kk + kgB];
        acc = __builtin_amdgcn_wmma_f32_16x16x32_f16(
                  /*neg_a=*/false, a, /*neg_b=*/false, b,
                  /*c_mod=*/(short)0, acc, /*reuse_a=*/false, /*reuse_b=*/false);
    }

    const int mbase = hiHalf * 8;
    #pragma unroll
    for (int r = 0; r < 8; ++r) {
        int orow = tile_m * 16 + mbase + r;
        if (orow < N)
            h[(long long)orow * Hdim + tile_n * 16 + n] = acc[r];
    }
}

// One wave32 per edge: metadata loads are wave-broadcast (single cache req),
// each lane moves 2 features: one b64 gather + two f32 atomics.
__global__ void __launch_bounds__(256)
k_scatter(const long long* __restrict__ src,
          const long long* __restrict__ dst,
          const float* __restrict__ ew,
          const float* __restrict__ dinv,
          const float* __restrict__ h,
          float* __restrict__ agg, int E) {
    const int lane = threadIdx.x & 31;
    const int wave = threadIdx.x >> 5;
    long long e = (long long)blockIdx.x * 8 + wave;
    if (e >= E) return;                   // wave-uniform
    int s = (int)src[e];
    int d = (int)dst[e];
    float w = dinv[s] * ew[e] * dinv[d];
    const float2* hs = (const float2*)(h + (long long)s * Hdim);
    float2 v = hs[lane];                  // features 2*lane, 2*lane+1
    float* ad = agg + (long long)d * Hdim + 2 * lane;
    atomicAdd(ad,     v.x * w);
    atomicAdd(ad + 1, v.y * w);
}

// relu(agg + self-loop + b1), block-local LDS reduction, then S atomics.
__global__ void __launch_bounds__(256)
k_node_reduce(const float* __restrict__ agg, const float* __restrict__ h,
              const float* __restrict__ dinv, const float* __restrict__ b1,
              float* __restrict__ S, int N) {
    long long idx = (long long)blockIdx.x * blockDim.x + threadIdx.x;
    int i = (int)(idx >> 6);
    int j = (int)(idx & 63);
    float v = 0.0f;
    if (i < N) {
        float di = dinv[i];
        long long o = (long long)i * Hdim + j;
        v = fmaxf(agg[o] + h[o] * di * di + b1[j], 0.0f);
    }
    __shared__ float sh[Hdim];
    if (threadIdx.x < Hdim) sh[threadIdx.x] = 0.0f;
    __syncthreads();
    atomicAdd(&sh[j], v);                 // ds_add_f32
    __syncthreads();
    if (threadIdx.x < Hdim) atomicAdd(&S[threadIdx.x], sh[threadIdx.x]);
}

// logits[j] = sum_k S[k]*W2[k,j] + N*b2[j]; res = log_softmax(logits)
__global__ void k_final(const float* __restrict__ S, const float* __restrict__ W2,
                        const float* __restrict__ b2, float* __restrict__ res, int N) {
    __shared__ float logits[Cdim];
    __shared__ float mx, lse;
    int j = threadIdx.x;
    if (j < Cdim) {
        float acc = (float)N * b2[j];
        #pragma unroll
        for (int k = 0; k < Hdim; ++k)
            acc += S[k] * W2[k * Cdim + j];
        logits[j] = acc;
    }
    __syncthreads();
    if (j == 0) {
        float m = -INFINITY;
        for (int t = 0; t < Cdim; ++t) m = fmaxf(m, logits[t]);
        float s = 0.0f;
        for (int t = 0; t < Cdim; ++t) s += expf(logits[t] - m);
        mx = m; lse = logf(s);
    }
    __syncthreads();
    if (j < Cdim) res[j] = logits[j] - mx - lse;
}

__global__ void k_bcast(const float* __restrict__ res, float* __restrict__ out,
                        long long total) {
    long long idx = (long long)blockIdx.x * blockDim.x + threadIdx.x;
    if (idx >= total) return;
    out[idx] = res[idx % Cdim];
}

extern "C" void kernel_launch(void* const* d_in, const int* in_sizes, int n_in,
                              void* d_out, int out_size, void* d_ws, size_t ws_size,
                              hipStream_t stream) {
    const float*     x   = (const float*)d_in[0];
    const long long* ei  = (const long long*)d_in[1];   // int64 [2,E]
    const float*     ew  = (const float*)d_in[2];
    const float*     W1  = (const float*)d_in[3];       // [128,64]
    const float*     b1  = (const float*)d_in[4];       // [64]
    const float*     W2  = (const float*)d_in[5];       // [64,40]
    const float*     b2  = (const float*)d_in[6];       // [40]
    float*           out = (float*)d_out;

    const int N = in_sizes[0] / Fin;
    const int E = in_sizes[1] / 2;
    const long long* src = ei;
    const long long* dst = ei + E;

    // workspace carve-up (floats): deg[N] dinv[N] h[N*64] agg[N*64] S[64] res[40]
    float* deg  = (float*)d_ws;
    float* dinv = deg  + N;
    float* h    = dinv + N;
    float* agg  = h    + (long long)N * Hdim;
    float* S    = agg  + (long long)N * Hdim;
    float* res  = S    + Hdim;

    const int T = 256;
    long long nh = (long long)N * Hdim;

    k_init<<<(int)((nh + T - 1) / T), T, 0, stream>>>(deg, agg, S, N);
    k_deg<<<(E + T - 1) / T, T, 0, stream>>>(dst, ew, deg, E);
    k_dinv<<<(N + T - 1) / T, T, 0, stream>>>(deg, dinv, N);

    int tiles  = ((N + 15) / 16) * (Hdim / 16);         // waves needed
    int blocks = (tiles + 7) / 8;                       // 8 wave32 / 256-thr block
    k_gemm_wmma<<<blocks, T, 0, stream>>>(x, W1, h, N);

    k_scatter<<<(E + 7) / 8, T, 0, stream>>>(src, dst, ew, dinv, h, agg, E);

    k_node_reduce<<<(int)((nh + T - 1) / T), T, 0, stream>>>(agg, h, dinv, b1, S, N);
    k_final<<<1, 64, 0, stream>>>(S, W2, b2, res, N);

    long long total = (long long)N * Cdim;
    k_bcast<<<(int)((total + T - 1) / T), T, 0, stream>>>(res, out, total);
}
